// GAT_76287209111908
// MI455X (gfx1250) — matrix operand order repaired and checked by reading, not compiled, not run on hardware
//
#include <hip/hip_runtime.h>
#include <hip/hip_bf16.h>
#include <math.h>

typedef __attribute__((ext_vector_type(16))) _Float16 v16h;
typedef __attribute__((ext_vector_type(8)))  float    v8f;

#define F_IN       512
#define HEADS      8
#define HID        8
#define C1         (HEADS*HID)   /* 64 */
#define NCLS       7
#define NEG_SLOPE  0.2f
#define MT         4             /* M tiles (16 rows each) per wave */

// ---------------------------------------------------------------------------
// helpers
// ---------------------------------------------------------------------------
__device__ __forceinline__ float leaky(float v) { return v > 0.f ? v : NEG_SLOPE * v; }

// f32 atomic max via integer ordering trick (init value must be -inf)
__device__ __forceinline__ void atomicMaxF(float* addr, float v) {
    if (v >= 0.f) atomicMax((int*)addr, __float_as_int(v));
    else          atomicMin((unsigned int*)addr, __float_as_uint(v));
}

__global__ void k_fill(float* __restrict__ p, float v, int n) {
    int i = blockIdx.x * blockDim.x + threadIdx.x;
    if (i < n) p[i] = v;
}

// ---------------------------------------------------------------------------
// GEMM1: h1[N,64] = x[N,512] @ W1[512,64]   (f32 -> f16 WMMA, f32 accumulate)
// block = 256 threads = 8 wave32s: 2 wave-rows (64 rows each) x 4 col-tiles.
// Block tile = 128 x 64. W1 pre-swizzled in LDS into the exact B-fragment
// layout so each wave reads one aligned v16h (2x ds_load_b128) per K-step,
// reused by MT=4 WMMAs. All MT A-fragments are built before the WMMA burst so
// consecutive WMMAs have disjoint A/acc registers (no per-WMMA hazard v_nops).
// ---------------------------------------------------------------------------
__global__ __launch_bounds__(256) void k_gemm1(const float* __restrict__ x,
                                               const float* __restrict__ W1,
                                               float* __restrict__ h1, int N) {
    // sB[ks][wn][lane][j]: 16 * 4 * 32 * 16 halves = 64 KB
    __shared__ __align__(128) _Float16 sB[16 * 4 * 32 * 16];
    const int tid = threadIdx.x;

    // fill pre-swizzled B: fragment element j of lane l in col-tile wn, k-step ks
// holds W1[k][col] with k = ks*32 + (l>=16)*8 + (j<8 ? j : j+8), col = wn*16 + (l&15)
    for (int g = tid; g < 4096; g += 256) {        // 8 halves per group
        const int j0   = (g & 1) * 8;
        const int lane = (g >> 1) & 31;
        const int wn   = (g >> 6) & 3;
        const int ks   = g >> 8;
        const int kb   = ks * 32 + (lane >> 4) * 8 + j0 * 2;
        const int col  = wn * 16 + (lane & 15);
        _Float16* dst = sB + (((ks * 4 + wn) * 32 + lane) * 16 + j0);
        #pragma unroll
        for (int j = 0; j < 8; ++j) dst[j] = (_Float16)W1[(kb + j) * C1 + col];
    }
    __syncthreads();

    const int wave = tid >> 5;
    const int lane = tid & 31;
    const int wn   = wave & 3;
    const int n0   = wn * 16;
    const int rowBase = blockIdx.x * 128 + (wave >> 2) * 64;  // this wave: 64 rows

    // A-fragment lane mapping (16-bit A 16x32): row = lane&15,
    //   lane<16 -> K in {0..7} U {16..23}; lane>=16 -> +8
    const int am   = lane & 15;
    const int aklo = (lane >> 4) * 8;

    const float* xr[MT];
    #pragma unroll
    for (int t = 0; t < MT; ++t) {
        int r = rowBase + t * 16 + am;
        if (r >= N) r = N - 1;                     // clamp: EXEC stays all-1 for WMMA
        xr[t] = x + (size_t)r * F_IN;
    }

    v8f acc[MT] = {};
    #pragma unroll 2
    for (int ks = 0; ks < 16; ++ks) {
        const int kb = ks * 32;
        // B fragment: one aligned 32-byte LDS read, shared by all MT WMMAs
        const v16h b = *(const v16h*)(sB + ((ks * 4 + wn) * 32 + lane) * 16);
        // build ALL A fragments first (keeps them in disjoint registers)
        v16h a[MT];
        #pragma unroll
        for (int t = 0; t < MT; ++t) {
            const float4 f0 = *(const float4*)(xr[t] + kb + aklo);
            const float4 f1 = *(const float4*)(xr[t] + kb + aklo + 4);
            const float4 f2 = *(const float4*)(xr[t] + kb + aklo + 16);
            const float4 f3 = *(const float4*)(xr[t] + kb + aklo + 20);
            a[t][0]=(_Float16)f0.x;  a[t][1]=(_Float16)f0.y;  a[t][2]=(_Float16)f0.z;  a[t][3]=(_Float16)f0.w;
            a[t][4]=(_Float16)f1.x;  a[t][5]=(_Float16)f1.y;  a[t][6]=(_Float16)f1.z;  a[t][7]=(_Float16)f1.w;
            a[t][8]=(_Float16)f2.x;  a[t][9]=(_Float16)f2.y;  a[t][10]=(_Float16)f2.z; a[t][11]=(_Float16)f2.w;
            a[t][12]=(_Float16)f3.x; a[t][13]=(_Float16)f3.y; a[t][14]=(_Float16)f3.z; a[t][15]=(_Float16)f3.w;
        }
        // WMMA burst: disjoint A and acc registers between consecutive ops
        #pragma unroll
        for (int t = 0; t < MT; ++t)
            acc[t] = __builtin_amdgcn_wmma_f32_16x16x32_f16(false, a[t], false, b,
                                                            (short)0, acc[t], false, false);
    }

    // C/D layout: VGPR r -> M = r + (lane>=16 ? 8 : 0), N = lane & 15
    const int mhi = (lane >> 4) * 8;
    const int col = n0 + (lane & 15);
    if (rowBase + 64 <= N) {                       // fast path: straight-line stores
        #pragma unroll
        for (int t = 0; t < MT; ++t) {
            float* hp = h1 + (size_t)(rowBase + t * 16 + mhi) * C1 + col;
            #pragma unroll
            for (int r = 0; r < 8; ++r) hp[(size_t)r * C1] = acc[t][r];
        }
    } else {                                       // ragged tail block
        #pragma unroll
        for (int t = 0; t < MT; ++t)
            #pragma unroll
            for (int r = 0; r < 8; ++r) {
                int row = rowBase + t * 16 + mhi + r;
                if (row < N) h1[(size_t)row * C1 + col] = acc[t][r];
            }
    }
}

// ---------------------------------------------------------------------------
// per-node attention halves, layer 1: al[n,h] = sum_c h1[n,h,c] * a[h,c]
// ---------------------------------------------------------------------------
__global__ void k_att1(const float* __restrict__ h1,
                       const float* __restrict__ a_src, const float* __restrict__ a_dst,
                       float* __restrict__ als, float* __restrict__ ald, int N) {
    int t = blockIdx.x * blockDim.x + threadIdx.x;  // t = node*8 + head
    if (t >= N * HEADS) return;
    int h = t & 7;
    const float* hp = h1 + (size_t)(t >> 3) * C1 + h * HID;
    float s = 0.f, d = 0.f;
    #pragma unroll
    for (int c = 0; c < HID; ++c) { s += hp[c] * a_src[h*HID+c]; d += hp[c] * a_dst[h*HID+c]; }
    als[t] = s; ald[t] = d;
}

// ---------------------------------------------------------------------------
// edge pipeline, layer 1 (HEADS channels): self-loops appended past E edges
// ---------------------------------------------------------------------------
__device__ __forceinline__ void edge_sd(const int* __restrict__ ei, int e, int E,
                                        int& s, int& d) {
    if (e < E) { s = ei[e]; d = ei[E + e]; } else { s = e - E; d = e - E; }
}

__global__ void k_edge_max1(const int* __restrict__ ei, const float* __restrict__ als,
                            const float* __restrict__ ald, float* __restrict__ m1,
                            int E, int ET) {
    int t = blockIdx.x * blockDim.x + threadIdx.x;  // t = edge*8 + head
    if (t >= ET * HEADS) return;
    int e = t >> 3, h = t & 7, s, d;
    edge_sd(ei, e, E, s, d);
    atomicMaxF(&m1[d * HEADS + h], leaky(als[s * HEADS + h] + ald[d * HEADS + h]));
}

__global__ void k_edge_sum1(const int* __restrict__ ei, const float* __restrict__ als,
                            const float* __restrict__ ald, const float* __restrict__ m1,
                            float* __restrict__ dn1, int E, int ET) {
    int t = blockIdx.x * blockDim.x + threadIdx.x;
    if (t >= ET * HEADS) return;
    int e = t >> 3, h = t & 7, s, d;
    edge_sd(ei, e, E, s, d);
    float v = leaky(als[s * HEADS + h] + ald[d * HEADS + h]);
    atomicAdd(&dn1[d * HEADS + h], __expf(v - m1[d * HEADS + h]));
}

__global__ void k_aggr1(const int* __restrict__ ei, const float* __restrict__ als,
                        const float* __restrict__ ald, const float* __restrict__ m1,
                        const float* __restrict__ dn1, const float* __restrict__ h1,
                        float* __restrict__ o1, int E, int ET) {
    int t = blockIdx.x * blockDim.x + threadIdx.x;  // t = edge*64 + feature
    if (t >= ET * C1) return;
    int e = t >> 6, f = t & 63, h = f >> 3, s, d;
    edge_sd(ei, e, E, s, d);
    float v = leaky(als[s * HEADS + h] + ald[d * HEADS + h]);
    float alpha = __expf(v - m1[d * HEADS + h]) / (dn1[d * HEADS + h] + 1e-16f);
    atomicAdd(&o1[(size_t)d * C1 + f], alpha * h1[(size_t)s * C1 + f]);
}

__global__ void k_elu1(float* __restrict__ o1, const float* __restrict__ b1, int N) {
    int t = blockIdx.x * blockDim.x + threadIdx.x;
    if (t >= N * C1) return;
    float v = o1[t] + b1[t & 63];
    o1[t] = v > 0.f ? v : (__expf(v) - 1.f);
}

// ---------------------------------------------------------------------------
// layer 2: h2[n,7] = elu(o1)[n,:] @ W2[64,7]; fused attention-half dots
// ---------------------------------------------------------------------------
__global__ __launch_bounds__(256) void k_l2(const float* __restrict__ o1,
                                            const float* __restrict__ W2,
                                            const float* __restrict__ a2s,
                                            const float* __restrict__ a2d,
                                            float* __restrict__ h2,
                                            float* __restrict__ als2,
                                            float* __restrict__ ald2, int N) {
    __shared__ float sW[C1 * NCLS], sas[NCLS], sad[NCLS];
    int tid = threadIdx.x;
    for (int i = tid; i < C1 * NCLS; i += 256) sW[i] = W2[i];
    if (tid < NCLS) { sas[tid] = a2s[tid]; sad[tid] = a2d[tid]; }
    __syncthreads();
    int n = blockIdx.x * 256 + tid;
    if (n >= N) return;
    const float* hp = o1 + (size_t)n * C1;
    float acc[NCLS] = {};
    for (int c = 0; c < C1; ++c) {
        float xv = hp[c];
        #pragma unroll
        for (int j = 0; j < NCLS; ++j) acc[j] += xv * sW[c * NCLS + j];
    }
    float s = 0.f, d = 0.f;
    #pragma unroll
    for (int j = 0; j < NCLS; ++j) {
        h2[(size_t)n * 8 + j] = acc[j];
        s += acc[j] * sas[j]; d += acc[j] * sad[j];
    }
    als2[n] = s; ald2[n] = d;
}

// ---------------------------------------------------------------------------
// edge pipeline, layer 2 (single head, 7 channels)
// ---------------------------------------------------------------------------
__global__ void k_edge_max2(const int* __restrict__ ei, const float* __restrict__ als,
                            const float* __restrict__ ald, float* __restrict__ m2,
                            int E, int ET) {
    int e = blockIdx.x * blockDim.x + threadIdx.x;
    if (e >= ET) return;
    int s, d; edge_sd(ei, e, E, s, d);
    atomicMaxF(&m2[d], leaky(als[s] + ald[d]));
}

__global__ void k_edge_sum2(const int* __restrict__ ei, const float* __restrict__ als,
                            const float* __restrict__ ald, const float* __restrict__ m2,
                            float* __restrict__ dn2, int E, int ET) {
    int e = blockIdx.x * blockDim.x + threadIdx.x;
    if (e >= ET) return;
    int s, d; edge_sd(ei, e, E, s, d);
    atomicAdd(&dn2[d], __expf(leaky(als[s] + ald[d]) - m2[d]));
}

__global__ void k_aggr2(const int* __restrict__ ei, const float* __restrict__ als,
                        const float* __restrict__ ald, const float* __restrict__ m2,
                        const float* __restrict__ dn2, const float* __restrict__ h2,
                        float* __restrict__ o2, int E, int ET) {
    int t = blockIdx.x * blockDim.x + threadIdx.x;  // t = edge*8 + channel
    if (t >= ET * 8) return;
    int e = t >> 3, c = t & 7;
    if (c >= NCLS) return;
    int s, d; edge_sd(ei, e, E, s, d);
    float v = leaky(als[s] + ald[d]);
    float alpha = __expf(v - m2[d]) / (dn2[d] + 1e-16f);
    atomicAdd(&o2[(size_t)d * 8 + c], alpha * h2[(size_t)s * 8 + c]);
}

__global__ void k_logsoftmax(const float* __restrict__ o2, const float* __restrict__ b2,
                             float* __restrict__ out, int N) {
    int n = blockIdx.x * blockDim.x + threadIdx.x;
    if (n >= N) return;
    float v[NCLS], m = -INFINITY;
    #pragma unroll
    for (int c = 0; c < NCLS; ++c) { v[c] = o2[(size_t)n * 8 + c] + b2[c]; m = fmaxf(m, v[c]); }
    float se = 0.f;
    #pragma unroll
    for (int c = 0; c < NCLS; ++c) se += __expf(v[c] - m);
    float lse = m + logf(se);
    #pragma unroll
    for (int c = 0; c < NCLS; ++c) out[(size_t)n * NCLS + c] = v[c] - lse;
}

// ---------------------------------------------------------------------------
// launch
// ---------------------------------------------------------------------------
extern "C" void kernel_launch(void* const* d_in, const int* in_sizes, int n_in,
                              void* d_out, int out_size, void* d_ws, size_t ws_size,
                              hipStream_t stream) {
    (void)n_in; (void)out_size; (void)ws_size;
    const float* x   = (const float*)d_in[0];
    const int*   ei  = (const int*)  d_in[1];
    const float* W1  = (const float*)d_in[2];
    const float* a1s = (const float*)d_in[3];
    const float* a1d = (const float*)d_in[4];
    const float* b1  = (const float*)d_in[5];
    const float* W2  = (const float*)d_in[6];
    const float* a2s = (const float*)d_in[7];
    const float* a2d = (const float*)d_in[8];
    const float* b2  = (const float*)d_in[9];
    float* out = (float*)d_out;

    const int N  = in_sizes[0] / F_IN;
    const int E  = in_sizes[1] / 2;
    const int ET = E + N;

    // workspace carve (floats): ~36 MB total for N=50000
    float* ws   = (float*)d_ws;
    float* h1   = ws;  ws += (size_t)N * C1;
    float* o1   = ws;  ws += (size_t)N * C1;
    float* als1 = ws;  ws += (size_t)N * HEADS;
    float* ald1 = ws;  ws += (size_t)N * HEADS;
    float* m1   = ws;  ws += (size_t)N * HEADS;
    float* dn1  = ws;  ws += (size_t)N * HEADS;
    float* h2   = ws;  ws += (size_t)N * 8;
    float* als2 = ws;  ws += (size_t)N;
    float* ald2 = ws;  ws += (size_t)N;
    float* m2   = ws;  ws += (size_t)N;
    float* dn2  = ws;  ws += (size_t)N;
    float* o2   = ws;  ws += (size_t)N * 8;

    const int B = 256;
    #define GRID(n) dim3(((n) + B - 1) / B)

    // re-init accumulators every call (graph replay must be deterministic)
    k_fill<<<GRID(N*HEADS), B, 0, stream>>>(m1, -INFINITY, N*HEADS);
    k_fill<<<GRID(N*HEADS), B, 0, stream>>>(dn1, 0.f,       N*HEADS);
    k_fill<<<GRID(N*C1),    B, 0, stream>>>(o1, 0.f,        N*C1);
    k_fill<<<GRID(N),       B, 0, stream>>>(m2, -INFINITY,  N);
    k_fill<<<GRID(N),       B, 0, stream>>>(dn2, 0.f,       N);
    k_fill<<<GRID(N*8),     B, 0, stream>>>(o2, 0.f,        N*8);

    // layer 1
    k_gemm1<<<dim3((N + 127) / 128), B, 0, stream>>>(x, W1, h1, N);
    k_att1<<<GRID(N*HEADS), B, 0, stream>>>(h1, a1s, a1d, als1, ald1, N);
    k_edge_max1<<<GRID(ET*HEADS), B, 0, stream>>>(ei, als1, ald1, m1, E, ET);
    k_edge_sum1<<<GRID(ET*HEADS), B, 0, stream>>>(ei, als1, ald1, m1, dn1, E, ET);
    k_aggr1<<<GRID(ET*C1), B, 0, stream>>>(ei, als1, ald1, m1, dn1, h1, o1, E, ET);
    k_elu1<<<GRID(N*C1), B, 0, stream>>>(o1, b1, N);

    // layer 2
    k_l2<<<GRID(N), B, 0, stream>>>(o1, W2, a2s, a2d, h2, als2, ald2, N);
    k_edge_max2<<<GRID(ET), B, 0, stream>>>(ei, als2, ald2, m2, E, ET);
    k_edge_sum2<<<GRID(ET), B, 0, stream>>>(ei, als2, ald2, m2, dn2, E, ET);
    k_aggr2<<<GRID(ET*8), B, 0, stream>>>(ei, als2, ald2, m2, dn2, h2, o2, E, ET);
    k_logsoftmax<<<GRID(N), B, 0, stream>>>(o2, b2, out, N);
    #undef GRID
}
